// LSTMFeatureExtractor_73100343378283
// MI455X (gfx1250) — compile-verified, matrix operand order checked by reading, not run on hardware
//
#include <hip/hip_runtime.h>

// ---------------------------------------------------------------------------
// Fused LSTM feature extractor for MI455X (gfx1250, wave32, WMMA).
// B=2048, T=1024, D=32, H=10.  One wave32 handles a 16-row batch tile and
// runs the whole T=1024 recurrence; gates computed with v_wmma_f32_16x16x32_f16
// (K=32 matches D exactly; h is zero-padded 10->32 in K).
// Activations use v_exp_f32 + v_rcp_f32 (fast rcp) to keep the serial
// per-timestep critical path short.
// ---------------------------------------------------------------------------

typedef _Float16 half_t;
typedef __attribute__((ext_vector_type(16))) _Float16 v16h;
typedef __attribute__((ext_vector_type(8)))  float    v8f;

#define B_SZ 2048
#define T_SZ 1024
#define D_SZ 32
#define H_SZ 10
#define G4   40   // 4*H gate columns (padded to 48 = 3 WMMA N-tiles)

__device__ __forceinline__ float fast_rcp(float x) {
    return __builtin_amdgcn_rcpf(x);          // single v_rcp_f32
}
__device__ __forceinline__ float sigm_f(float x) {
    return fast_rcp(1.0f + __expf(-x));       // v_exp_f32 + v_add + v_rcp
}
__device__ __forceinline__ float tanh_f(float x) {
    float e = __expf(-2.0f * x);
    return (1.0f - e) * fast_rcp(1.0f + e);
}

__global__ __launch_bounds__(32)
void lstm_fused_wmma_kernel(const float* __restrict__ x,      // [B,T,D]
                            const float* __restrict__ W_ih,   // [4H,D]
                            const float* __restrict__ W_hh,   // [4H,H]
                            const float* __restrict__ b_ih,   // [4H]
                            const float* __restrict__ b_hh,   // [4H]
                            float*       __restrict__ out)    // [B,H]
{
    // LDS staging: gates in [col][row] so a lane's 8 C/D rows are contiguous.
    __shared__ float  gLds[48][20];      // padded rows (16B-aligned, spread banks)
    __shared__ half_t hLds[16][16];      // h as row-major f16, cols >= H stay zero

    const int lane = threadIdx.x;        // 0..31, wave32 == workgroup
    const int ln15 = lane & 15;          // A-operand row / C,D column index
    const int sub  = lane >> 4;          // half-wave selector in the 16-bit layouts
    const int b0   = blockIdx.x * 16;    // batch tile base

    // ---- pack W_ih / W_hh into WMMA B-operand layout (16-bit, 32x16 per tile) ----
    // B element j of the per-lane v16h maps to K = (j>=8 ? 16 : 0) + sub*8 + (j&7),
    // column N = tile*16 + ln15.  gates = A @ W^T  =>  B[K][N] = W[N][K].
    v16h  wihB[3], whhB[3];
    float biasc[3];
#pragma unroll
    for (int ct = 0; ct < 3; ++ct) {
        const int n = ct * 16 + ln15;
#pragma unroll
        for (int j = 0; j < 16; ++j) {
            const int K = ((j >= 8) ? 16 : 0) + sub * 8 + (j & 7);
            float wv = (n < G4) ? W_ih[n * D_SZ + K] : 0.0f;
            wihB[ct][j] = (half_t)wv;
            float hv = (n < G4 && K < H_SZ) ? W_hh[n * H_SZ + K] : 0.0f;
            whhB[ct][j] = (half_t)hv;
        }
        biasc[ct] = (n < G4) ? (b_ih[n] + b_hh[n]) : 0.0f;
    }

    // ---- zero h LDS (keeps the padded columns zero forever) ----
#pragma unroll
    for (int q = 0; q < 8; ++q)
        ((half_t*)hLds)[lane * 8 + q] = (half_t)0.0f;
    __syncthreads();

    // ---- per-lane recurrent state: 5 of the 160 (row, j) cells ----
    int rr[5], jj[5];
#pragma unroll
    for (int q = 0; q < 5; ++q) {
        const int e = lane * 5 + q;
        rr[q] = e / 10;
        jj[q] = e % 10;
    }
    float cst[5]  = {0, 0, 0, 0, 0};
    float hsum[5] = {0, 0, 0, 0, 0};
    v16h  hA = {};                        // h in A-operand layout; starts at zero

    // x pointer for this lane's A-operand slice: row b0+ln15, d base = sub*8.
    const float* xrow = x + ((size_t)(b0 + ln15) * T_SZ) * D_SZ + sub * 8;

    for (int t = 0; t < T_SZ; ++t) {
        // -------- load x tile as A operand (4 x global_load_b128 per lane) --------
        float4 a0 = *(const float4*)(xrow);
        float4 a1 = *(const float4*)(xrow + 4);
        float4 a2 = *(const float4*)(xrow + 16);
        float4 a3 = *(const float4*)(xrow + 20);
        __builtin_prefetch(xrow + D_SZ, 0, 0);   // next timestep (global_prefetch_b8)

        v16h xA;
        xA[0]  = (half_t)a0.x; xA[1]  = (half_t)a0.y; xA[2]  = (half_t)a0.z; xA[3]  = (half_t)a0.w;
        xA[4]  = (half_t)a1.x; xA[5]  = (half_t)a1.y; xA[6]  = (half_t)a1.z; xA[7]  = (half_t)a1.w;
        xA[8]  = (half_t)a2.x; xA[9]  = (half_t)a2.y; xA[10] = (half_t)a2.z; xA[11] = (half_t)a2.w;
        xA[12] = (half_t)a3.x; xA[13] = (half_t)a3.y; xA[14] = (half_t)a3.z; xA[15] = (half_t)a3.w;

        // -------- gates[16 x 48] = bias + x@W_ih^T + h@W_hh^T via WMMA --------
#pragma unroll
        for (int ct = 0; ct < 3; ++ct) {
            v8f acc;
#pragma unroll
            for (int r = 0; r < 8; ++r) acc[r] = biasc[ct];
            acc = __builtin_amdgcn_wmma_f32_16x16x32_f16(
                      false, xA, false, wihB[ct], (short)0, acc, false, false);
            acc = __builtin_amdgcn_wmma_f32_16x16x32_f16(
                      false, hA, false, whhB[ct], (short)0, acc, false, false);
            // C/D layout: this lane holds column ct*16+ln15, rows sub*8 .. sub*8+7.
            float* gp = &gLds[ct * 16 + ln15][sub * 8];
            *(float4*)(gp)     = make_float4(acc[0], acc[1], acc[2], acc[3]);
            *(float4*)(gp + 4) = make_float4(acc[4], acc[5], acc[6], acc[7]);
        }
        __syncthreads();   // single-wave WG: lowers to dscnt wait / s_nop

        // -------- pointwise LSTM cell update on 5 cells per lane --------
#pragma unroll
        for (int q = 0; q < 5; ++q) {
            const int r2 = rr[q], j2 = jj[q];
            const float gi = gLds[j2][r2];
            const float gf = gLds[10 + j2][r2];
            const float gg = gLds[20 + j2][r2];
            const float go = gLds[30 + j2][r2];
            const float iv = sigm_f(gi);
            const float fv = sigm_f(gf);
            const float gv = tanh_f(gg);
            const float ov = sigm_f(go);
            const float cn = fv * cst[q] + iv * gv;
            cst[q] = cn;
            const float hv = ov * tanh_f(cn);
            hsum[q] += hv;
            hLds[r2][j2] = (half_t)hv;
        }
        __syncthreads();

        // -------- rebuild h in A-operand layout (one 16B LDS read; K>=16 is zero) --------
        float4 hraw = *(const float4*)&hLds[ln15][sub * 8];
        const half_t* hp = (const half_t*)&hraw;
#pragma unroll
        for (int j = 0; j < 8; ++j) hA[j] = hp[j];
#pragma unroll
        for (int j = 8; j < 16; ++j) hA[j] = (half_t)0.0f;

        xrow += D_SZ;
    }

    // -------- write mean over time: out[b, j] --------
    const float inv_t = 1.0f / (float)T_SZ;
#pragma unroll
    for (int q = 0; q < 5; ++q)
        out[(size_t)(b0 + rr[q]) * H_SZ + jj[q]] = hsum[q] * inv_t;
}

extern "C" void kernel_launch(void* const* d_in, const int* in_sizes, int n_in,
                              void* d_out, int out_size, void* d_ws, size_t ws_size,
                              hipStream_t stream) {
    (void)in_sizes; (void)n_in; (void)d_ws; (void)ws_size; (void)out_size;
    const float* x    = (const float*)d_in[0];
    const float* W_ih = (const float*)d_in[1];
    const float* W_hh = (const float*)d_in[2];
    const float* b_ih = (const float*)d_in[3];
    const float* b_hh = (const float*)d_in[4];
    float* out = (float*)d_out;

    dim3 grid(B_SZ / 16);   // 128 batch tiles, one wave32 each
    dim3 block(32);
    lstm_fused_wmma_kernel<<<grid, block, 0, stream>>>(x, W_ih, W_hh, b_ih, b_hh, out);
}
